// GraphBertAutoencoder_42271068127546
// MI455X (gfx1250) — matrix operand order repaired and checked by reading, not compiled, not run on hardware
//
#include <hip/hip_runtime.h>

typedef __attribute__((ext_vector_type(16))) _Float16 v16h;
typedef __attribute__((ext_vector_type(8)))  _Float16 v8h;
typedef __attribute__((ext_vector_type(8)))  float    v8f;
typedef __attribute__((ext_vector_type(2)))  float    v2f;

static const int kN = 65536, kB = 128, kMAXN = 512, kE = 1048576;
static const int kEMBED = 768, kHID = 256, kLAT = 64;

#define TBM 256          // rows per block
#define TBN 64           // cols per block
#define TBK 32           // k per iteration
#define SB_STRIDE 40     // sBt row stride in halves (80B, keeps 16B alignment for b128)

// ---------------- LayerNorm over EMBED=768, one wave per row ----------------
__global__ __launch_bounds__(256) void ln_kernel(const float* __restrict__ x,
    const float* __restrict__ gamma, const float* __restrict__ beta,
    float* __restrict__ out)
{
  int row  = blockIdx.x * 8 + (threadIdx.x >> 5);
  int lane = threadIdx.x & 31;
  const float* xr = x + (size_t)row * kEMBED;
  float s = 0.f, s2 = 0.f;
  for (int c = lane; c < kEMBED; c += 32) { float v = xr[c]; s += v; s2 += v * v; }
  for (int off = 16; off > 0; off >>= 1) { s += __shfl_xor(s, off); s2 += __shfl_xor(s2, off); }
  float mu  = s * (1.f / kEMBED);
  float inv = rsqrtf(s2 * (1.f / kEMBED) - mu * mu + 1e-5f);
  float* o = out + (size_t)row * kEMBED;
  for (int c = lane; c < kEMBED; c += 32) o[c] = (xr[c] - mu) * inv * gamma[c] + beta[c];
}

// ---------------- degree / norm ----------------
__global__ void deg_init_kernel(float* __restrict__ deg) {
  deg[(size_t)blockIdx.x * blockDim.x + threadIdx.x] = 1.0f;   // self-loop
}
__global__ void deg_count_kernel(const int* __restrict__ dst, float* __restrict__ deg) {
  int e = blockIdx.x * blockDim.x + threadIdx.x;
  atomicAdd(&deg[dst[e]], 1.0f);
}
__global__ void dinv_kernel(const float* __restrict__ deg, float* __restrict__ dinv) {
  int i = blockIdx.x * blockDim.x + threadIdx.x;
  dinv[i] = rsqrtf(deg[i]);
}

// ---------------- weight fp32 -> fp16 transpose: Wt[Co][K] = (half)W[K][Co] ----------------
__global__ void wtrans_kernel(const float* __restrict__ W, _Float16* __restrict__ Wt,
                              int K, int Co)
{
  int k = blockIdx.x;
  for (int c = threadIdx.x; c < Co; c += 256)
    Wt[(size_t)c * K + k] = (_Float16)W[(size_t)k * Co + c];
}

// ---------------- fp16-WMMA GEMM: C[N,Co] = relu?(A[N,K]) @ W[K,Co] (+bias) ----------------
// block = 256 threads (8 waves); block tile 256x64; wave tile 32x64 (8 accumulators).
// A tile: global f32 -> (relu) -> f16 -> LDS. B tile: async copy of pre-transposed fp16
// weights straight into LDS (global_load_async_to_lds_b128, ASYNCcnt-tracked).
__global__ __launch_bounds__(256) void gemm_kernel(const float* __restrict__ A,
    const _Float16* __restrict__ Wt, const float* __restrict__ bias,
    float* __restrict__ C, int K, int Co, int doRelu)
{
  __shared__ _Float16 sA[TBM][TBK];            // 16 KB
  __shared__ _Float16 sBt[TBN][SB_STRIDE];     // 5 KB, transposed: [col][k]
  const int rowBase = blockIdx.x * TBM;
  const int colBase = blockIdx.y * TBN;
  const int tid  = threadIdx.x;
  const int wave = tid >> 5;
  const int lane = tid & 31;
  const int m0   = wave * 32;                  // wave's 32-row stripe
  const int ml   = lane & 15;
  const int hi   = lane >> 4;
  v8f acc[2][4] = {};

  const int bn  = tid >> 2;                    // B fill: col 0..63
  const int bs  = (tid & 3) * 8;               // B fill: k segment (halves)
  const uint32_t ldsB = (uint32_t)(uintptr_t)(&sBt[bn][bs]);

  for (int kk = 0; kk < K; kk += TBK) {
    // ---- B tile: async global(fp16, pre-transposed) -> LDS, 16B per lane ----
    uint64_t gaddr = (uint64_t)(uintptr_t)(Wt + (size_t)(colBase + bn) * K + kk + bs);
    asm volatile("global_load_async_to_lds_b128 %0, %1, off"
                 :: "v"(ldsB), "v"(gaddr) : "memory");

    // ---- A tile: one 32-wide k-slice of one row per thread, relu fused ----
    const float* ap = A + (size_t)(rowBase + tid) * K + kk;
#pragma unroll
    for (int i = 0; i < 32; ++i) {
      float v = ap[i];
      if (doRelu) v = v > 0.f ? v : 0.f;
      sA[tid][i] = (_Float16)v;
    }
    asm volatile("s_wait_asynccnt 0x0" ::: "memory");
    __syncthreads();

    // ---- fragments: all contiguous 16B LDS loads ----
    v16h af[2];
#pragma unroll
    for (int mb = 0; mb < 2; ++mb) {
      const _Float16* pa = &sA[m0 + mb * 16 + ml][hi * 16];
      v8h lo  = *(const v8h*)(pa);        // K = hi*16 .. +7
      v8h hi8 = *(const v8h*)(pa + 8);    // K = hi*16+8 .. +15
      af[mb] = __builtin_shufflevector(lo, hi8,
               0, 1, 2, 3, 4, 5, 6, 7, 8, 9, 10, 11, 12, 13, 14, 15);
    }
#pragma unroll
    for (int t = 0; t < 4; ++t) {
      const _Float16* pb = &sBt[t * 16 + ml][hi * 16];
      v16h bf = *(const v16h*)pb;         // K = hi*16 .. +15, contiguous
      acc[0][t] = __builtin_amdgcn_wmma_f32_16x16x32_f16(
          false, af[0], false, bf, (short)0, acc[0][t], false, false);
      acc[1][t] = __builtin_amdgcn_wmma_f32_16x16x32_f16(
          false, af[1], false, bf, (short)0, acc[1][t], false, false);
    }
    __syncthreads();
  }

#pragma unroll
  for (int mb = 0; mb < 2; ++mb) {
#pragma unroll
    for (int t = 0; t < 4; ++t) {
#pragma unroll
      for (int r = 0; r < 8; ++r) {
        int row = rowBase + m0 + mb * 16 + r + hi * 8;
        int col = colBase + t * 16 + ml;
        float v = acc[mb][t][r];
        if (bias) v += bias[col];
        C[(size_t)row * Co + col] = v;
      }
    }
  }
}

// ---------------- aggregation: out = H*dinv^2 (self loop) + bias, then edge atomics ----------------
__global__ __launch_bounds__(256) void agg_init_kernel(const float* __restrict__ H,
    const float* __restrict__ dinv, const float* __restrict__ bias,
    float* __restrict__ out, int coShift, int coMask)
{
  size_t i = (size_t)blockIdx.x * blockDim.x + threadIdx.x;
  int node = (int)(i >> coShift);
  int c    = (int)(i & (size_t)coMask);
  float di = dinv[node];
  out[i] = H[i] * di * di + bias[c];
}

__global__ __launch_bounds__(256) void scatter_kernel(const int* __restrict__ src,
    const int* __restrict__ dst, const float* __restrict__ H,
    const float* __restrict__ dinv, float* __restrict__ out,
    int Co, int co4Shift, int co4Mask)
{
  size_t t = (size_t)blockIdx.x * blockDim.x + threadIdx.x;
  int e  = (int)(t >> co4Shift);
  int c4 = ((int)(t & (size_t)co4Mask)) << 2;
  int s = src[e], d = dst[e];
  float w = dinv[s] * dinv[d];
  const float4 h = *(const float4*)(H + (size_t)s * Co + c4);   // L2-resident gather
  float* op = out + (size_t)d * Co + c4;
  atomicAdd(op + 0, h.x * w);
  atomicAdd(op + 1, h.y * w);
  atomicAdd(op + 2, h.z * w);
  atomicAdd(op + 3, h.w * w);
}

// ---------------- global mean pool (contiguous equal-size graphs) ----------------
__global__ void pool_kernel(const float* __restrict__ z, float* __restrict__ gemb) {
  int b = blockIdx.x;
  int c = threadIdx.x;                     // 64 threads = LAT
  const float* zp = z + (size_t)b * kMAXN * kLAT + c;
  float s = 0.f;
  for (int i = 0; i < kMAXN; ++i) s += zp[(size_t)i * kLAT];
  gemb[b * kLAT + c] = s * (1.f / kMAXN);
}

__global__ void mask_kernel(float* __restrict__ m) {
  m[(size_t)blockIdx.x * blockDim.x + threadIdx.x] = 1.0f;     // dense equal-size batch
}

// ---------------- pred_adj: sigmoid(Zb @ Zb^T) with fp32 WMMA 16x16x4 ----------------
__global__ __launch_bounds__(256) void adj_kernel(const float* __restrict__ z,
                                                  float* __restrict__ adj)
{
  int wave = threadIdx.x >> 5, lane = threadIdx.x & 31;
  int tile = blockIdx.x * 8 + wave;        // 128 graphs * 32*32 tiles
  int g  = tile >> 10;
  int t  = tile & 1023;
  int ti = t >> 5, tj = t & 31;
  const float* Z = z + (size_t)g * kMAXN * kLAT;
  int mrow = ti * 16 + (lane & 15);
  int ncol = tj * 16 + (lane & 15);
  int koff = (lane >> 4) * 2;              // A/B 16x4 f32 layout: lanes 16-31 hold K=2,3
  v8f acc = {};
#pragma unroll
  for (int k0 = 0; k0 < kLAT; k0 += 4) {
    v2f a, b;
    a.x = Z[(size_t)mrow * kLAT + k0 + koff];
    a.y = Z[(size_t)mrow * kLAT + k0 + koff + 1];
    b.x = Z[(size_t)ncol * kLAT + k0 + koff];   // B = Z^T => column n is row ncol of Z
    b.y = Z[(size_t)ncol * kLAT + k0 + koff + 1];
    acc = __builtin_amdgcn_wmma_f32_16x16x4_f32(
        false, a, false, b, (short)0, acc, false, false);
  }
  size_t base = (size_t)g * kMAXN * kMAXN;
#pragma unroll
  for (int r = 0; r < 8; ++r) {
    int row = ti * 16 + r + (lane >> 4) * 8;
    int col = tj * 16 + (lane & 15);
    adj[base + (size_t)row * kMAXN + col] = 1.f / (1.f + __expf(-acc[r]));
  }
}

// ---------------- host launcher ----------------
extern "C" void kernel_launch(void* const* d_in, const int* in_sizes, int n_in,
                              void* d_out, int out_size, void* d_ws, size_t ws_size,
                              hipStream_t stream)
{
  (void)in_sizes; (void)n_in; (void)out_size; (void)ws_size;
  const float* x     = (const float*)d_in[0];
  const int*   ei    = (const int*)d_in[1];
  const int*   src   = ei;
  const int*   dst   = ei + kE;
  const float* gamma = (const float*)d_in[3];
  const float* beta  = (const float*)d_in[4];
  const float* W1 = (const float*)d_in[5];  const float* b1 = (const float*)d_in[6];
  const float* W2 = (const float*)d_in[7];  const float* b2 = (const float*)d_in[8];
  const float* W3 = (const float*)d_in[9];  const float* b3 = (const float*)d_in[10];
  const float* W4 = (const float*)d_in[11]; const float* b4 = (const float*)d_in[12];
  const float* W5 = (const float*)d_in[13]; const float* b5 = (const float*)d_in[14];
  const float* Wd = (const float*)d_in[15]; const float* bd = (const float*)d_in[16];

  float* out = (float*)d_out;
  const size_t OFF_Z    = 0;
  const size_t OFF_GEMB = OFF_Z    + (size_t)kN * kLAT;
  const size_t OFF_ADJ  = OFF_GEMB + (size_t)kB * kLAT;
  const size_t OFF_MASK = OFF_ADJ  + (size_t)kB * kMAXN * kMAXN;
  const size_t OFF_LOG  = OFF_MASK + (size_t)kB * kMAXN;
  const size_t OFF_BERT = OFF_LOG  + (size_t)kN * kEMBED;
  float* zO    = out + OFF_Z;
  float* gembO = out + OFF_GEMB;
  float* adjO  = out + OFF_ADJ;
  float* maskO = out + OFF_MASK;
  float* logO  = out + OFF_LOG;
  float* bertO = out + OFF_BERT;

  // workspace: two N*256 f32 ping-pong buffers (each 67MB, both L2-resident),
  // deg/dinv, then fp16 transposed weights (~0.9 MB).
  float* H0   = (float*)d_ws;
  float* H1   = H0 + (size_t)kN * kHID;
  float* dinv = H1 + (size_t)kN * kHID;
  float* deg  = dinv + kN;
  _Float16* W1t = (_Float16*)(deg + kN);
  _Float16* W2t = W1t + (size_t)kEMBED * kHID;
  _Float16* W3t = W2t + (size_t)kHID * kHID;
  _Float16* W4t = W3t + (size_t)kHID * kHID;
  _Float16* W5t = W4t + (size_t)kHID * kHID;
  _Float16* Wdt = W5t + (size_t)kHID * kLAT;

  ln_kernel<<<kN / 8, 256, 0, stream>>>(x, gamma, beta, bertO);
  deg_init_kernel<<<kN / 256, 256, 0, stream>>>(deg);
  deg_count_kernel<<<kE / 256, 256, 0, stream>>>(dst, deg);
  dinv_kernel<<<kN / 256, 256, 0, stream>>>(deg, dinv);

  wtrans_kernel<<<kEMBED, 256, 0, stream>>>(W1, W1t, kEMBED, kHID);
  wtrans_kernel<<<kHID,   256, 0, stream>>>(W2, W2t, kHID,   kHID);
  wtrans_kernel<<<kHID,   256, 0, stream>>>(W3, W3t, kHID,   kHID);
  wtrans_kernel<<<kHID,   256, 0, stream>>>(W4, W4t, kHID,   kHID);
  wtrans_kernel<<<kHID,   256, 0, stream>>>(W5, W5t, kHID,   kLAT);
  wtrans_kernel<<<kLAT,   256, 0, stream>>>(Wd, Wdt, kLAT,   kEMBED);

  auto gcn = [&](const float* in, int K, const _Float16* Wt, const float* bias,
                 int Co, float* gemmOut, float* aggOut, bool reluIn) {
    dim3 g(kN / TBM, Co / TBN);
    gemm_kernel<<<g, 256, 0, stream>>>(in, Wt, nullptr, gemmOut, K, Co, reluIn ? 1 : 0);
    int coShift = (Co == 256) ? 8 : 6;
    agg_init_kernel<<<(unsigned)(((size_t)kN * Co) / 256), 256, 0, stream>>>(
        gemmOut, dinv, bias, aggOut, coShift, Co - 1);
    scatter_kernel<<<(unsigned)(((size_t)kE * (Co / 4)) / 256), 256, 0, stream>>>(
        src, dst, gemmOut, dinv, aggOut, Co, coShift - 2, (Co / 4) - 1);
  };

  // H1 always stores the pre-ReLU aggregated output; the consumer GEMM fuses ReLU.
  gcn(bertO, kEMBED, W1t, b1, kHID, H0, H1, false);
  gcn(H1,    kHID,   W2t, b2, kHID, H0, H1, true);
  gcn(H1,    kHID,   W3t, b3, kHID, H0, H1, true);
  gcn(H1,    kHID,   W4t, b4, kHID, H0, H1, true);
  gcn(H1,    kHID,   W5t, b5, kLAT, H0, zO, true);

  pool_kernel<<<kB, kLAT, 0, stream>>>(zO, gembO);
  mask_kernel<<<(kB * kMAXN) / 256, 256, 0, stream>>>(maskO);
  adj_kernel<<<(kB * 1024) / 8, 256, 0, stream>>>(zO, adjO);

  dim3 gl(kN / TBM, kEMBED / TBN);
  gemm_kernel<<<gl, 256, 0, stream>>>(zO, Wdt, bd, logO, kLAT, kEMBED, 0);
}
